// RelativeMultiHeadAttention_39548058861863
// MI455X (gfx1250) — matrix operand order repaired and checked by reading, not compiled, not run on hardware
//
#include <hip/hip_runtime.h>

typedef unsigned short u16;
typedef __attribute__((ext_vector_type(16))) __bf16 v16bf;
typedef __attribute__((ext_vector_type(8)))  float  v8f;
typedef int v4i_vs __attribute__((vector_size(16)));

#define BB 4
#define SS 1024
#define DD 512
#define HH 8
#define DHD 64
#define RPAD 192
#define RROWS (RPAD + SS) /* 1216 */

union BF16Frag { v16bf v; u16 s[16]; uint4 q[2]; };
union BF8Pack  { u16 s[8]; uint4 q; };

__device__ __forceinline__ u16 f2bf(float f) {
  unsigned u = __builtin_bit_cast(unsigned, f);
  unsigned r = (u + 0x7FFFu + ((u >> 16) & 1u)) >> 16;  // RNE
  return (u16)r;
}

__device__ __forceinline__ v8f wmma_bf16(v16bf a, v16bf b, v8f c) {
  return __builtin_amdgcn_wmma_f32_16x16x32_bf16(false, a, false, b, (short)0, c, false, false);
}

// B fragment: b[e] = B[k = e + 16*h2][n];  16 contiguous bf16 at p (32B aligned)
__device__ __forceinline__ v16bf load_bfrag(const u16* p) {
  BF16Frag f;
  f.q[0] = *reinterpret_cast<const uint4*>(p);
  f.q[1] = *reinterpret_cast<const uint4*>(p + 8);
  return f.v;
}

// A fragment: a[e] = row[kbase + e + 16*h2]
__device__ __forceinline__ v16bf load_afrag_bf16(const u16* rowp, int kbase, int h2) {
  BF16Frag f;
  f.q[0] = *reinterpret_cast<const uint4*>(rowp + kbase + 16 * h2);
  f.q[1] = *reinterpret_cast<const uint4*>(rowp + kbase + 16 * h2 + 8);
  return f.v;
}

__device__ __forceinline__ v16bf load_afrag_f32(const float* rowp, int kbase, int h2) {
  BF16Frag f;
#pragma unroll
  for (int e = 0; e < 8; ++e) f.s[e] = f2bf(rowp[kbase + 16 * h2 + e]);
#pragma unroll
  for (int e = 0; e < 8; ++e) f.s[8 + e] = f2bf(rowp[kbase + 16 * h2 + 8 + e]);
  return f.v;
}

// 16-byte async copy global -> LDS (ASYNCcnt-tracked, no VGPR data)
__device__ __forceinline__ void async_cp16(const u16* gptr, u16* lptr) {
  __builtin_amdgcn_global_load_async_to_lds_b128(
      (__attribute__((address_space(1))) v4i_vs*)(u16*)gptr,
      (__attribute__((address_space(3))) v4i_vs*)lptr, 0, 0);
}

// ---------------------------------------------------------------------------
// Weight transpose + bf16 convert: WT[n][k] = (bf16) W[k][n]
__global__ __launch_bounds__(256) void transpose_w_kernel(const float* __restrict__ W,
                                                          u16* __restrict__ WT) {
  int idx = blockIdx.x * 256 + threadIdx.x;
  if (idx >= DD * DD) return;
  int k = idx & (DD - 1), n = idx >> 9;
  WT[n * DD + k] = f2bf(W[k * DD + n]);
}

// RrT[h][k+RPAD][d] = rel_emb[h][d][S-1-k]  (zeros for k<0 pad region)
__global__ __launch_bounds__(256) void build_rrt_kernel(const float* __restrict__ rel,
                                                        u16* __restrict__ RrT) {
  int idx = blockIdx.x * 256 + threadIdx.x;
  if (idx >= HH * RROWS * DHD) return;
  int d = idx & 63;
  int kk = (idx >> 6) % RROWS;
  int h = idx / (RROWS * DHD);
  float v = 0.f;
  if (kk >= RPAD) {
    int k = kk - RPAD;
    v = rel[(h * DHD + d) * SS + (SS - 1 - k)];
  }
  RrT[idx] = f2bf(v);
}

// ---------------------------------------------------------------------------
// Tiled WMMA GEMM: out[4096x512] = A[4096x512] @ W, with W given transposed.
// OUTMODE: 0 = bf16 row-major, 1 = Vt bf16 [B*H][64][1024], 2 = f32 row-major
template <bool A_IS_F32, int OUTMODE>
__global__ __launch_bounds__(256) void proj_gemm(const void* __restrict__ Ap,
                                                 const u16* __restrict__ WT,
                                                 void* __restrict__ outp) {
  const int lane = threadIdx.x & 31;
  const int w = threadIdx.x >> 5;
  const int n16 = lane & 15;
  const int h2 = lane >> 4;
  const int r0 = blockIdx.x * 128 + w * 16;
  const int n0 = blockIdx.y * 64;

  v8f acc[4];
#pragma unroll
  for (int t = 0; t < 4; ++t) acc[t] = (v8f){0.f, 0.f, 0.f, 0.f, 0.f, 0.f, 0.f, 0.f};

  const int arow = r0 + n16;
#pragma unroll 4
  for (int kc = 0; kc < DD; kc += 32) {
    v16bf a;
    if (A_IS_F32) a = load_afrag_f32((const float*)Ap + (size_t)arow * DD, kc, h2);
    else          a = load_afrag_bf16((const u16*)Ap + (size_t)arow * DD, kc, h2);
#pragma unroll
    for (int t = 0; t < 4; ++t) {
      const u16* bp = WT + (size_t)(n0 + t * 16 + n16) * DD + kc + 16 * h2;
      acc[t] = wmma_bf16(a, load_bfrag(bp), acc[t]);
    }
  }

  if (OUTMODE == 0) {
    u16* out = (u16*)outp;
#pragma unroll
    for (int t = 0; t < 4; ++t) {
      int col = n0 + t * 16 + n16;
#pragma unroll
      for (int v = 0; v < 8; ++v)
        out[(size_t)(r0 + v + 8 * h2) * DD + col] = f2bf(acc[t][v]);
    }
  } else if (OUTMODE == 1) {
    u16* vt = (u16*)outp;
    int b = r0 >> 10;
    int s0 = (r0 & (SS - 1)) + 8 * h2;
#pragma unroll
    for (int t = 0; t < 4; ++t) {
      int col = n0 + t * 16 + n16;
      int h = col >> 6, dh = col & 63;
      BF8Pack pk;
#pragma unroll
      for (int v = 0; v < 8; ++v) pk.s[v] = f2bf(acc[t][v]);
      *reinterpret_cast<uint4*>(vt + (((size_t)(b * HH + h) * DHD + dh) * SS + s0)) = pk.q;
    }
  } else {
    float* out = (float*)outp;
#pragma unroll
    for (int t = 0; t < 4; ++t) {
      int col = n0 + t * 16 + n16;
#pragma unroll
      for (int v = 0; v < 8; ++v)
        out[(size_t)(r0 + v + 8 * h2) * DD + col] = acc[t][v];
    }
  }
}

// ---------------------------------------------------------------------------
// Fused flash attention with skewed relative bias.
// bias[i,j] = E[i, i-j], E = Q @ RrT^T (band computed per key tile via WMMA).
// K/V tiles are staged cooperatively through LDS with async copies.
__global__ __launch_bounds__(256) void rel_attn_kernel(const u16* __restrict__ Qb,
                                                       const u16* __restrict__ Kb,
                                                       const u16* __restrict__ Vt,
                                                       const u16* __restrict__ RrT,
                                                       u16* __restrict__ ctx) {
  // Per-wave scratch: E band (16x80 f32) reused later for P (16x64 bf16) —
  // disjoint lifetimes within a wave; same-wave LDS ops retire in order.
  __shared__ __align__(16) float SCR[8][16 * 80];        // 40960 B
  __shared__ __align__(16) u16 Ktile[64 * 64];           // 8192 B  [j][d]
  __shared__ __align__(16) u16 Vtile[64 * 64];           // 8192 B  [d][j]

  const int tid = threadIdx.x;
  const int lane = tid & 31;
  const int w = tid >> 5;
  const int n16 = lane & 15;
  const int h2 = lane >> 4;

  const int blk = blockIdx.x;
  const int mblk = blk & 7;   // S/128 blocks
  const int bh = blk >> 3;
  const int h = bh & 7;
  const int b = bh >> 3;

  const int i0 = mblk * 128;
  const int r0 = i0 + w * 16;  // this wave's query-row base (sequence index)

  const u16* Qrow  = Qb + (size_t)(b * SS + r0 + n16) * DD + h * DHD;
  const u16* Kbase = Kb + (size_t)(b * SS) * DD + h * DHD;
  const u16* Vbase = Vt + (size_t)bh * DHD * SS;
  const u16* Rr    = RrT + (size_t)h * RROWS * DHD;

  float* Ew = SCR[w];
  u16*   Pw = (u16*)SCR[w];

  v16bf aq0 = load_afrag_bf16(Qrow, 0, h2);
  v16bf aq1 = load_afrag_bf16(Qrow, 32, h2);

  const v8f vzero = (v8f){0.f, 0.f, 0.f, 0.f, 0.f, 0.f, 0.f, 0.f};
  v8f o[4];
  float m_run[8], l_run[8];
#pragma unroll
  for (int t = 0; t < 4; ++t) o[t] = vzero;
#pragma unroll
  for (int v = 0; v < 8; ++v) { m_run[v] = -1e30f; l_run[v] = 0.f; }

  const float scale = 0.125f;  // DH^-0.5
  const int jend = i0 + 128;

  for (int j0 = 0; j0 < jend; j0 += 64) {
    // ---- cooperative async stage of K tile [64j x 64d] and V tile [64d x 64j]
#pragma unroll
    for (int c = 0; c < 2; ++c) {
      const int idx = c * 256 + tid;         // 512 chunks of 8 u16
      const int row = idx >> 3;
      const int col8 = (idx & 7) * 8;
      async_cp16(Kbase + (size_t)(j0 + row) * DD + col8, &Ktile[row * 64 + col8]);
      async_cp16(Vbase + (size_t)row * SS + j0 + col8, &Vtile[row * 64 + col8]);
    }
    __builtin_amdgcn_s_wait_asynccnt(0);
    __syncthreads();

    if (j0 <= r0 + 15) {  // wave-uniform: skip tiles fully above the diagonal
      // ---- S = Q K^T  (16 x 64 strip), B fragments from LDS
      v8f s[4];
#pragma unroll
      for (int t = 0; t < 4; ++t) {
        const u16* kp = &Ktile[(t * 16 + n16) * 64 + 16 * h2];
        v8f acc = vzero;
        acc = wmma_bf16(aq0, load_bfrag(kp), acc);
        acc = wmma_bf16(aq1, load_bfrag(kp + 32), acc);
        s[t] = acc;
      }

      // ---- relative-bias band E[i, k], k=i-j, col = k-(r0-j0-64) in [0,80)
      const int kk0 = r0 - j0 + 128;  // kmin + RPAD
#pragma unroll
      for (int t = 0; t < 5; ++t) {
        const u16* rp = Rr + (size_t)(kk0 + t * 16 + n16) * DHD + 16 * h2;
        v8f e = vzero;
        e = wmma_bf16(aq0, load_bfrag(rp), e);
        e = wmma_bf16(aq1, load_bfrag(rp + 32), e);
#pragma unroll
        for (int v = 0; v < 8; ++v)
          Ew[(v + 8 * h2) * 80 + t * 16 + n16] = e[v];
      }

      // ---- logits = (qk + bias + causal_mask) * scale (diagonal gather of E)
#pragma unroll
      for (int t = 0; t < 4; ++t) {
        const int j = j0 + t * 16 + n16;
#pragma unroll
        for (int v = 0; v < 8; ++v) {
          const int m = v + 8 * h2;
          const int i = r0 + m;
          const float bias = Ew[m * 80 + (64 + m - t * 16 - n16)];
          const float mk = (j > i) ? -1e9f : 0.f;
          s[t][v] = (s[t][v] + bias + mk) * scale;
        }
      }

      // ---- online softmax (row reductions across 16-lane halves)
      float alpha[8];
#pragma unroll
      for (int v = 0; v < 8; ++v) {
        float mx = fmaxf(fmaxf(s[0][v], s[1][v]), fmaxf(s[2][v], s[3][v]));
#pragma unroll
        for (int msk = 1; msk < 16; msk <<= 1) mx = fmaxf(mx, __shfl_xor(mx, msk, 32));
        float mnew = fmaxf(m_run[v], mx);
        alpha[v] = __expf(m_run[v] - mnew);
        m_run[v] = mnew;
      }
#pragma unroll
      for (int v = 0; v < 8; ++v) {
        float rs = 0.f;
#pragma unroll
        for (int t = 0; t < 4; ++t) {
          float p = __expf(s[t][v] - m_run[v]);
          s[t][v] = p;
          rs += p;
        }
#pragma unroll
        for (int msk = 1; msk < 16; msk <<= 1) rs += __shfl_xor(rs, msk, 32);
        l_run[v] = l_run[v] * alpha[v] + rs;
      }

      // ---- P (C-layout) -> LDS -> A-fragment layout (reuses E scratch)
#pragma unroll
      for (int t = 0; t < 4; ++t)
#pragma unroll
        for (int v = 0; v < 8; ++v)
          Pw[(v + 8 * h2) * 64 + t * 16 + n16] = f2bf(s[t][v]);

      v16bf ap0 = load_afrag_bf16(&Pw[n16 * 64], 0, h2);
      v16bf ap1 = load_afrag_bf16(&Pw[n16 * 64], 32, h2);

      // ---- O = O*alpha + P @ V  (V tile transposed in LDS)
#pragma unroll
      for (int t = 0; t < 4; ++t) {
#pragma unroll
        for (int v = 0; v < 8; ++v) o[t][v] *= alpha[v];
        const u16* vp = &Vtile[(t * 16 + n16) * 64 + 16 * h2];
        o[t] = wmma_bf16(ap0, load_bfrag(vp), o[t]);
        o[t] = wmma_bf16(ap1, load_bfrag(vp + 32), o[t]);
      }

      if (j0 + 64 < jend)
        __builtin_prefetch(Kbase + (size_t)(j0 + 64 + n16 * 4) * DD, 0, 1);
    }

    __syncthreads();  // all waves done with K/V tiles before next overwrite
  }

  // ---- normalize, store ctx as bf16 [B,S,H*DH]
#pragma unroll
  for (int v = 0; v < 8; ++v) l_run[v] = 1.f / l_run[v];
#pragma unroll
  for (int t = 0; t < 4; ++t) {
    const int col = h * DHD + t * 16 + n16;
#pragma unroll
    for (int v = 0; v < 8; ++v) {
      const int srow = r0 + v + 8 * h2;
      ctx[(size_t)(b * SS + srow) * DD + col] = f2bf(o[t][v] * l_run[v]);
    }
  }
}

// ---------------------------------------------------------------------------
extern "C" void kernel_launch(void* const* d_in, const int* in_sizes, int n_in,
                              void* d_out, int out_size, void* d_ws, size_t ws_size,
                              hipStream_t stream) {
  (void)in_sizes; (void)n_in; (void)out_size; (void)ws_size;
  const float* queries = (const float*)d_in[0];
  const float* keysp   = (const float*)d_in[1];
  const float* values  = (const float*)d_in[2];
  // d_in[3] = mask: causal, synthesized analytically in-kernel
  const float* Wq = (const float*)d_in[4];
  const float* Wk = (const float*)d_in[5];
  const float* Wv = (const float*)d_in[6];
  const float* Wo = (const float*)d_in[7];
  const float* rel = (const float*)d_in[8];

  size_t off = 0;
  auto walloc = [&](size_t bytes) {
    void* p = (char*)d_ws + off;
    off += (bytes + 255) & ~(size_t)255;
    return p;
  };
  u16* Qb   = (u16*)walloc((size_t)BB * SS * DD * 2);
  u16* Kbuf = (u16*)walloc((size_t)BB * SS * DD * 2);
  u16* Vtb  = (u16*)walloc((size_t)BB * SS * DD * 2);
  u16* ctx  = (u16*)walloc((size_t)BB * SS * DD * 2);
  u16* WqT  = (u16*)walloc((size_t)DD * DD * 2);
  u16* WkT  = (u16*)walloc((size_t)DD * DD * 2);
  u16* WvT  = (u16*)walloc((size_t)DD * DD * 2);
  u16* WoT  = (u16*)walloc((size_t)DD * DD * 2);
  u16* RrT  = (u16*)walloc((size_t)HH * RROWS * DHD * 2);

  const int wgrid = (DD * DD + 255) / 256;
  transpose_w_kernel<<<wgrid, 256, 0, stream>>>(Wq, WqT);
  transpose_w_kernel<<<wgrid, 256, 0, stream>>>(Wk, WkT);
  transpose_w_kernel<<<wgrid, 256, 0, stream>>>(Wv, WvT);
  transpose_w_kernel<<<wgrid, 256, 0, stream>>>(Wo, WoT);
  build_rrt_kernel<<<(HH * RROWS * DHD + 255) / 256, 256, 0, stream>>>(rel, RrT);

  dim3 pg(32, 8);
  proj_gemm<true, 0><<<pg, 256, 0, stream>>>(queries, WqT, Qb);
  proj_gemm<true, 0><<<pg, 256, 0, stream>>>(keysp, WkT, Kbuf);
  proj_gemm<true, 1><<<pg, 256, 0, stream>>>(values, WvT, Vtb);

  rel_attn_kernel<<<BB * HH * (SS / 128), 256, 0, stream>>>(Qb, Kbuf, Vtb, RrT, ctx);

  proj_gemm<false, 2><<<pg, 256, 0, stream>>>(ctx, WoT, d_out);
}
